// ELRec_DLRM_Net_29901562314962
// MI455X (gfx1250) — matrix operand order repaired and checked by reading, not compiled, not run on hardware
//
#include <hip/hip_runtime.h>

typedef __attribute__((ext_vector_type(16))) _Float16 v16h;
typedef __attribute__((ext_vector_type(8)))  float    v8f;

constexpr int B_    = 16384;
constexpr int NTAB  = 26;
constexpr int NROWS = 100000;
constexpr int D_    = 64;

// ---------------------------------------------------------------------------
// f32 -> f16 conversion with zero column padding (src rows x sc -> dst rows x dc)
// ---------------------------------------------------------------------------
__global__ void cvt_pad_f16(const float* __restrict__ src, _Float16* __restrict__ dst,
                            int rows, int sc, int dc) {
  int idx = blockIdx.x * 256 + threadIdx.x;
  if (idx >= rows * dc) return;
  int r = idx / dc;
  int c = idx - r * dc;
  dst[idx] = (c < sc) ? (_Float16)src[(size_t)r * sc + c] : (_Float16)0.f;
}

// ---------------------------------------------------------------------------
// C = relu(A (M x K, f16) * W^T (N x K, f16) + bias), f32 accumulate via WMMA.
// Block tile: (WM*64) x (WN*32), 256 threads = 8 waves, each wave 4x2 tiles.
// Double-buffered LDS: global loads for K-step k+1 are issued before the
// WMMAs of step k so HBM/L2 latency hides behind the matrix pipe.
// K must be a multiple of 32.
// ---------------------------------------------------------------------------
template <int WM, int WN>
__global__ __launch_bounds__(256) void gemm_bias_relu_f16(
    const _Float16* __restrict__ A, const _Float16* __restrict__ Wt,
    const float* __restrict__ bias, _Float16* __restrict__ C,
    int N, int K) {
  constexpr int BM   = WM * 64;
  constexpr int BN   = WN * 32;
  constexpr int AREG = (BM * 4) / 256;          // uint4 per thread for A tile
  constexpr int WREG = (BN * 4 + 255) / 256;    // uint4 per thread for W tile
  __shared__ __align__(16) _Float16 As[2][BM][32];
  __shared__ __align__(16) _Float16 Ws[2][BN][32];

  const int tid  = threadIdx.x;
  const int lane = tid & 31;
  const int wid  = tid >> 5;
  const int wm   = wid % WM;
  const int wn   = wid / WM;
  const int m0   = blockIdx.x * BM;
  const int n0   = blockIdx.y * BN;
  const int r    = lane & 15;   // N index inside a C/D tile; M row group inside A/B
  const int hi   = lane >> 4;   // lane half-group

  uint4 areg[AREG];
  uint4 wreg[WREG];

  auto gload = [&](int k0) {
#pragma unroll
    for (int i = 0; i < AREG; ++i) {
      int q = tid + i * 256, row = q >> 2, seg = q & 3;
      areg[i] = *reinterpret_cast<const uint4*>(&A[(size_t)(m0 + row) * K + k0 + seg * 8]);
    }
#pragma unroll
    for (int i = 0; i < WREG; ++i) {
      int q = tid + i * 256;
      if (q < BN * 4) {
        int row = q >> 2, seg = q & 3;
        wreg[i] = *reinterpret_cast<const uint4*>(&Wt[(size_t)(n0 + row) * K + k0 + seg * 8]);
      }
    }
  };
  auto lstore = [&](int buf) {
#pragma unroll
    for (int i = 0; i < AREG; ++i) {
      int q = tid + i * 256, row = q >> 2, seg = q & 3;
      *reinterpret_cast<uint4*>(&As[buf][row][seg * 8]) = areg[i];
    }
#pragma unroll
    for (int i = 0; i < WREG; ++i) {
      int q = tid + i * 256;
      if (q < BN * 4) {
        int row = q >> 2, seg = q & 3;
        *reinterpret_cast<uint4*>(&Ws[buf][row][seg * 8]) = wreg[i];
      }
    }
  };

  v8f zero = {};
  v8f acc[4][2];
#pragma unroll
  for (int mi = 0; mi < 4; ++mi)
#pragma unroll
    for (int ni = 0; ni < 2; ++ni) acc[mi][ni] = zero;

  gload(0);
  lstore(0);
  const int nk = K >> 5;
  for (int kk = 0; kk < nk; ++kk) {
    __syncthreads();
    const int cur = kk & 1;
    if (kk + 1 < nk) gload((kk + 1) << 5);   // in-flight while we do WMMA

    v16h afrag[4], bfrag[2];
#pragma unroll
    for (int mi = 0; mi < 4; ++mi) {  // 16-bit A 16x32 layout: lane group hi covers K = hi*8..+7, hi*8+16..+23
      const int row = wm * 64 + mi * 16 + r;
      union { v16h v; uint4 q[2]; } u;
      u.q[0] = *reinterpret_cast<const uint4*>(&As[cur][row][hi * 8]);
      u.q[1] = *reinterpret_cast<const uint4*>(&As[cur][row][hi * 8 + 16]);
      afrag[mi] = u.v;
    }
#pragma unroll
    for (int ni = 0; ni < 2; ++ni) {  // 16-bit B 32x16 layout: lane group hi covers K = hi*8..+7, hi*8+16..+23
      const int row = wn * 32 + ni * 16 + r;
      union { v16h v; uint4 q[2]; } u;
      u.q[0] = *reinterpret_cast<const uint4*>(&Ws[cur][row][hi * 8]);
      u.q[1] = *reinterpret_cast<const uint4*>(&Ws[cur][row][hi * 8 + 16]);
      bfrag[ni] = u.v;
    }
#pragma unroll
    for (int mi = 0; mi < 4; ++mi)
#pragma unroll
      for (int ni = 0; ni < 2; ++ni)
        acc[mi][ni] = __builtin_amdgcn_wmma_f32_16x16x32_f16(
            false, afrag[mi], false, bfrag[ni], (short)0, acc[mi][ni], false, false);

    if (kk + 1 < nk) lstore(cur ^ 1);
  }

  // epilogue: bias + relu + f16 store.  C/D layout: N = lane&15, M = reg + 8*hi
#pragma unroll
  for (int ni = 0; ni < 2; ++ni) {
    const int n  = n0 + wn * 32 + ni * 16 + r;
    const float bv = bias[n];
#pragma unroll
    for (int mi = 0; mi < 4; ++mi) {
      const int mbase = m0 + wm * 64 + mi * 16 + hi * 8;
#pragma unroll
      for (int j = 0; j < 8; ++j) {
        float v = acc[mi][ni][j] + bv;
        v = v > 0.f ? v : 0.f;
        C[(size_t)(mbase + j) * N + n] = (_Float16)v;
      }
    }
  }
}

// ---------------------------------------------------------------------------
// Fused embedding gather + pairwise interaction + R assembly.
// One wave per sample (8 samples / workgroup).  All 26 indices are read first
// and every embedding row is prefetched (global_prefetch_b8) so the 26 random
// 256B HBM fetches are in flight concurrently instead of serially.
// T (27x64, padded to 32x64, f16) lives in LDS; Z = T*T^T via 2x2x2 WMMA
// tiles; lower triangle scattered straight from accumulators into
// R[b] = [x(64) | Zflat(351) | 0].
// ---------------------------------------------------------------------------
__global__ __launch_bounds__(256) void interact_kernel(
    const _Float16* __restrict__ X2, const int* __restrict__ lSi,
    const float* __restrict__ emb, _Float16* __restrict__ R) {
  __shared__ __align__(16) _Float16 Tsh[8][32][64];
  const int lane = threadIdx.x & 31;
  const int w    = threadIdx.x >> 5;
  const int b    = blockIdx.x * 8 + w;

  // gather indices, then prefetch all 26 rows (each lane covers 8B of the row)
  int idxs[NTAB];
#pragma unroll
  for (int t = 0; t < NTAB; ++t) {
    idxs[t] = lSi[t * B_ + b];
    __builtin_prefetch(emb + ((size_t)t * NROWS + idxs[t]) * D_ + lane * 2, 0, 3);
  }

  // row 0 = bottom-MLP output (already f16): 32 lanes x 2 halves
  {
    const unsigned int* px = reinterpret_cast<const unsigned int*>(X2 + (size_t)b * D_);
    *reinterpret_cast<unsigned int*>(&Tsh[w][0][lane * 2]) = px[lane];
  }
  // rows 1..26 = gathered embedding rows (f32 -> f16)
#pragma unroll
  for (int t = 0; t < NTAB; ++t) {
    const float2* src = reinterpret_cast<const float2*>(emb + ((size_t)t * NROWS + idxs[t]) * D_);
    float2 v = src[lane];
    Tsh[w][1 + t][lane * 2]     = (_Float16)v.x;
    Tsh[w][1 + t][lane * 2 + 1] = (_Float16)v.y;
  }
#pragma unroll
  for (int row = 27; row < 32; ++row)
    *reinterpret_cast<unsigned int*>(&Tsh[w][row][lane * 2]) = 0u;
  __syncthreads();

  const int r  = lane & 15;
  const int hi = lane >> 4;
  v8f zero = {};
  v8f acc[2][2];
#pragma unroll
  for (int mi = 0; mi < 2; ++mi)
#pragma unroll
    for (int ni = 0; ni < 2; ++ni) acc[mi][ni] = zero;

#pragma unroll
  for (int kb = 0; kb < 2; ++kb) {
    v16h af[2], bf[2];
#pragma unroll
    for (int mi = 0; mi < 2; ++mi) {
      const int row = mi * 16 + r;
      union { v16h v; uint4 q[2]; } u;
      u.q[0] = *reinterpret_cast<const uint4*>(&Tsh[w][row][kb * 32 + hi * 8]);
      u.q[1] = *reinterpret_cast<const uint4*>(&Tsh[w][row][kb * 32 + hi * 8 + 16]);
      af[mi] = u.v;
    }
#pragma unroll
    for (int ni = 0; ni < 2; ++ni) {
      const int row = ni * 16 + r;
      union { v16h v; uint4 q[2]; } u;
      u.q[0] = *reinterpret_cast<const uint4*>(&Tsh[w][row][kb * 32 + hi * 8]);
      u.q[1] = *reinterpret_cast<const uint4*>(&Tsh[w][row][kb * 32 + hi * 8 + 16]);
      bf[ni] = u.v;
    }
#pragma unroll
    for (int mi = 0; mi < 2; ++mi)
#pragma unroll
      for (int ni = 0; ni < 2; ++ni)
        acc[mi][ni] = __builtin_amdgcn_wmma_f32_16x16x32_f16(
            false, af[mi], false, bf[ni], (short)0, acc[mi][ni], false, false);
  }

  // R[b][0..63] = x
  {
    const unsigned int* px = reinterpret_cast<const unsigned int*>(&Tsh[w][0][0]);
    *reinterpret_cast<unsigned int*>(&R[(size_t)b * 416 + lane * 2]) = px[lane];
  }
  if (lane == 0) R[(size_t)b * 416 + 415] = (_Float16)0.f;  // pad column

  // strict lower triangle of Z (27x27), row-major tril order: p = m(m-1)/2 + n
#pragma unroll
  for (int mi = 0; mi < 2; ++mi)
#pragma unroll
    for (int ni = 0; ni < 2; ++ni)
#pragma unroll
      for (int j = 0; j < 8; ++j) {
        const int m = mi * 16 + hi * 8 + j;
        const int n = ni * 16 + r;
        if (m <= 26 && n < m)
          R[(size_t)b * 416 + 64 + (m * (m - 1)) / 2 + n] = (_Float16)acc[mi][ni][j];
      }
}

// ---------------------------------------------------------------------------
// Final 256 -> 1 layer + sigmoid (N=1: plain per-row dot product)
// ---------------------------------------------------------------------------
__global__ void top_final_kernel(const _Float16* __restrict__ Y1,
                                 const float* __restrict__ w2,
                                 const float* __restrict__ b2,
                                 float* __restrict__ out) {
  int row = blockIdx.x * 256 + threadIdx.x;
  if (row >= B_) return;
  const _Float16* y = Y1 + (size_t)row * 256;
  float acc = b2[0];
#pragma unroll 8
  for (int k = 0; k < 256; ++k) acc += (float)y[k] * w2[k];
  out[row] = 1.0f / (1.0f + __expf(-acc));
}

// ---------------------------------------------------------------------------
extern "C" void kernel_launch(void* const* d_in, const int* in_sizes, int n_in,
                              void* d_out, int out_size, void* d_ws, size_t ws_size,
                              hipStream_t stream) {
  (void)in_sizes; (void)n_in; (void)out_size; (void)ws_size;
  const float* dense_x = (const float*)d_in[0];
  const int*   lSi     = (const int*)d_in[1];
  const float* emb     = (const float*)d_in[2];
  const float* bw0 = (const float*)d_in[3];  const float* bb0 = (const float*)d_in[4];
  const float* bw1 = (const float*)d_in[5];  const float* bb1 = (const float*)d_in[6];
  const float* bw2 = (const float*)d_in[7];  const float* bb2 = (const float*)d_in[8];
  const float* tw0 = (const float*)d_in[9];  const float* tb0 = (const float*)d_in[10];
  const float* tw1 = (const float*)d_in[11]; const float* tb1 = (const float*)d_in[12];
  const float* tw2 = (const float*)d_in[13]; const float* tb2 = (const float*)d_in[14];
  float* out = (float*)d_out;

  char* p = (char*)d_ws;
  auto take = [&](size_t halves) -> _Float16* {
    _Float16* q = (_Float16*)p;
    p += (halves * sizeof(_Float16) + 255) & ~(size_t)255;
    return q;
  };
  _Float16* dxp  = take((size_t)B_ * 32);   // dense_x padded 13->32
  _Float16* w_b0 = take(512 * 32);          // bot_w0 padded 13->32
  _Float16* w_b1 = take(256 * 512);
  _Float16* w_b2 = take(64 * 256);
  _Float16* w_t0 = take(512 * 416);         // top_w0 padded 415->416
  _Float16* w_t1 = take(256 * 512);
  _Float16* X0   = take((size_t)B_ * 512);
  _Float16* X1   = take((size_t)B_ * 256);
  _Float16* X2   = take((size_t)B_ * 64);
  _Float16* R    = take((size_t)B_ * 416);
  _Float16* Y0   = X0;  // X0 dead after X1
  _Float16* Y1   = X1;  // X1 dead after X2

  auto cvt = [&](const float* s, _Float16* d, int rows, int sc, int dc) {
    int total = rows * dc;
    cvt_pad_f16<<<dim3((total + 255) / 256), dim3(256), 0, stream>>>(s, d, rows, sc, dc);
  };
  cvt(dense_x, dxp, B_, 13, 32);
  cvt(bw0, w_b0, 512, 13, 32);
  cvt(bw1, w_b1, 256, 512, 512);
  cvt(bw2, w_b2, 64, 256, 256);
  cvt(tw0, w_t0, 512, 415, 416);
  cvt(tw1, w_t1, 256, 512, 512);

  // bottom MLP
  gemm_bias_relu_f16<2, 4><<<dim3(B_ / 128, 512 / 128), 256, 0, stream>>>(dxp, w_b0, bb0, X0, 512, 32);
  gemm_bias_relu_f16<2, 4><<<dim3(B_ / 128, 256 / 128), 256, 0, stream>>>(X0, w_b1, bb1, X1, 256, 512);
  gemm_bias_relu_f16<4, 2><<<dim3(B_ / 256, 1),         256, 0, stream>>>(X1, w_b2, bb2, X2, 64, 256);
  // gather + interaction + R assembly
  interact_kernel<<<dim3(B_ / 8), 256, 0, stream>>>(X2, lSi, emb, R);
  // top MLP
  gemm_bias_relu_f16<2, 4><<<dim3(B_ / 128, 512 / 128), 256, 0, stream>>>(R, w_t0, tb0, Y0, 512, 416);
  gemm_bias_relu_f16<2, 4><<<dim3(B_ / 128, 256 / 128), 256, 0, stream>>>(Y0, w_t1, tb1, Y1, 256, 512);
  top_final_kernel<<<dim3(B_ / 256), 256, 0, stream>>>(Y1, tw2, tb2, out);
}
